// RNNEmbedding_91122026152234
// MI455X (gfx1250) — matrix operand order repaired and checked by reading, not compiled, not run on hardware
//
#include <hip/hip_runtime.h>

typedef __attribute__((ext_vector_type(16))) _Float16 v16h;
typedef __attribute__((ext_vector_type(8)))  float    v8f;

#define T_LEN 512
#define BATCH 64
#define IN0   256
#define HID   512
#define OUTD  128
#define TB    (T_LEN * BATCH)

#define SCAN_WGS     8      // workgroups per direction (M-split 4 x N-split 32 = 128 tiles)
#define SCAN_THREADS 512    // 16 waves -> one 16x16 tile per wave

// ---- gfx1250 async global->LDS staging (guarded) ---------------------------

#if defined(__has_builtin)
#if __has_builtin(__builtin_amdgcn_global_load_async_to_lds_b128) && \
    __has_builtin(__builtin_amdgcn_s_wait_asynccnt)
#define USE_ASYNC_LDS 1
#endif
#endif

#if defined(USE_ASYNC_LDS)
// builtin signature (from hipcc diagnostic): arg0 = global int4* (AS1, GCC vector),
// arg1 = LDS int4* (AS3), arg2 = imm offset, arg3 = cpol
typedef int v4i_gcc __attribute__((vector_size(16)));
typedef __attribute__((address_space(1))) v4i_gcc gv4i_t;
typedef __attribute__((address_space(3))) v4i_gcc lv4i_t;
#endif

// copy 8 halves (16 bytes) global -> LDS
__device__ __forceinline__ void stage16(const _Float16* g, _Float16* l) {
#if defined(USE_ASYNC_LDS)
    __builtin_amdgcn_global_load_async_to_lds_b128((gv4i_t*)g, (lv4i_t*)l, 0, 0);
#else
#pragma unroll
    for (int i = 0; i < 8; ++i) l[i] = g[i];
#endif
}

__device__ __forceinline__ void stage_wait() {
#if defined(USE_ASYNC_LDS)
    __builtin_amdgcn_s_wait_asynccnt(0);
#endif
}

// ---- WMMA helpers (CDNA5 16x16x32 f16 -> f32) ------------------------------

__device__ __forceinline__ v8f wmma_f16(v16h a, v16h b, v8f c) {
    return __builtin_amdgcn_wmma_f32_16x16x32_f16(false, a, false, b, (short)0, c,
                                                  false, false);
}

// A fragment (16x32 f16) from row-major storage (LDS or global), base pre-offset by K origin.
// lanes 0-15: M=lane, K=0..15 ; lanes 16-31: M=lane-16, K=16..31.
__device__ __forceinline__ v16h load_a_frag(const _Float16* base, int ld) {
    const int lane = threadIdx.x & 31;
    const int row  = lane & 15;
    const int kk   = (lane >> 4) << 4;
    const _Float16* p = base + row * ld + kk;
    v16h f;
#pragma unroll
    for (int i = 0; i < 16; ++i) f[i] = p[i];
    return f;
}

// B fragment (32x16 f16), B[k][n] = W[n][k], W row-major N x K, w pre-offset by k0.
__device__ __forceinline__ v16h load_b_frag(const _Float16* w, int ld, int n0) {
    const int lane = threadIdx.x & 31;
    const _Float16* p = w + (size_t)(n0 + (lane & 15)) * ld + ((lane >> 4) << 4);
    v16h f;
#pragma unroll
    for (int i = 0; i < 16; ++i) f[i] = p[i];
    return f;
}

// ---- f32 -> f16 conversion -------------------------------------------------

extern "C" __global__ void cvt_f16(const float* __restrict__ s,
                                   _Float16* __restrict__ d, int n) {
    const int i = blockIdx.x * blockDim.x + threadIdx.x;
    if (i < n) d[i] = (_Float16)s[i];
}

// ---- batched GEMM: C = A(MxK,f16) @ W(NxK,f16)^T + b0 + b1 (f32 out) -------

extern "C" __global__ void __launch_bounds__(256)
gemm_bias(const _Float16* __restrict__ A, int lda,
          const _Float16* __restrict__ W, int K,
          const float* __restrict__ b0, const float* __restrict__ b1,
          float* __restrict__ C, int ldc, int coloff) {
    __shared__ _Float16 sA[128 * 32];
    __shared__ _Float16 sB[64 * 32];

    const int tid  = threadIdx.x;
    const int bm   = blockIdx.x * 128;
    const int bn   = blockIdx.y * 64;
    const int wave = tid >> 5, lane = tid & 31;
    const int wm   = wave & 3, wn = wave >> 2;

    v8f acc[2][2] = {};

    for (int k0 = 0; k0 < K; k0 += 32) {
        {   // A chunk: 128 x 32 halves, 16/thread (2 async b128)
            const int r = tid >> 1, c = (tid & 1) * 16;
            const _Float16* src = A + (size_t)(bm + r) * lda + k0 + c;
            _Float16* dst = sA + r * 32 + c;
            stage16(src, dst);
            stage16(src + 8, dst + 8);
        }
        {   // W chunk: 64 x 32 halves, 8/thread (1 async b128)
            const int r = tid >> 2, c = (tid & 3) * 8;
            stage16(W + (size_t)(bn + r) * K + k0 + c, sB + r * 32 + c);
        }
        stage_wait();
        __syncthreads();
#pragma unroll
        for (int i2 = 0; i2 < 2; ++i2) {
            v16h a = load_a_frag(sA + (wm * 32 + i2 * 16) * 32, 32);
#pragma unroll
            for (int j = 0; j < 2; ++j) {
                v16h b = load_b_frag(sB, 32, wn * 32 + j * 16);
                acc[i2][j] = wmma_f16(a, b, acc[i2][j]);
            }
        }
        __syncthreads();
    }

#pragma unroll
    for (int i2 = 0; i2 < 2; ++i2)
#pragma unroll
        for (int j = 0; j < 2; ++j) {
            const int Nl = wn * 32 + j * 16 + (lane & 15);
            const int Ng = bn + Nl;
            const float bias = b0[Ng] + (b1 ? b1[Ng] : 0.f);
#pragma unroll
            for (int r = 0; r < 8; ++r) {
                const int M = bm + wm * 32 + i2 * 16 + r + ((lane >> 4) << 3);
                C[(size_t)M * ldc + coloff + Ng] = acc[i2][j][r] + bias;
            }
        }
}

// ---- scan init: zero barrier state, stage h0 (f16) -------------------------

extern "C" __global__ void scan_init(const float* __restrict__ h0,   // [2][B][H]
                                     _Float16* __restrict__ hbuf,    // [2][2][B*H]
                                     int* __restrict__ sync) {
    const int i = blockIdx.x * blockDim.x + threadIdx.x;
    if (i < 4) sync[i] = 0;
    if (i < 2 * BATCH * HID) {
        const int d = i / (BATCH * HID);
        const int r = i - d * (BATCH * HID);
        hbuf[(size_t)d * 2 * BATCH * HID + r] = (_Float16)h0[i];
    }
}

// ---- distributed recurrent scan --------------------------------------------
// grid = 2 dirs x SCAN_WGS blocks, 512 threads (16 waves). Wave owns one
// 16x16 output tile (m = batch tile, n = column tile); its Whh slice lives in
// VGPRs as 16 B-fragments for the whole scan. Hidden state ping-pongs through
// an L2-resident global f16 buffer; a per-direction sense-reversing software
// barrier (atomics + s_sleep) separates the 512 steps.

extern "C" __global__ void __launch_bounds__(SCAN_THREADS)
rnn_scan(const float* __restrict__ xw,       // [2][TB][HID]
         const _Float16* __restrict__ whh,   // [2][HID][HID]
         _Float16* hbuf,                      // [2][2][BATCH*HID] ping-pong
         _Float16* __restrict__ y,            // [TB][2*HID], col off = dir*HID
         float* __restrict__ hT,              // [2][BATCH][HID]
         int* sync)                           // [0..1]=cnt, [2..3]=gen
{
    const int dir  = blockIdx.x / SCAN_WGS;
    const bool rev = (dir == 1);
    const float*    xwd  = xw  + (size_t)dir * TB * HID;
    const _Float16* whhd = whh + (size_t)dir * HID * HID;
    float*          hTd  = hT  + (size_t)dir * BATCH * HID;
    _Float16*       hb0  = hbuf + (size_t)dir * 2 * BATCH * HID;
    _Float16*       hb1  = hb0 + BATCH * HID;
    int* cnt = sync + dir;
    int* gen = sync + 2 + dir;

    const int tid  = threadIdx.x;
    const int wave = tid >> 5;
    const int lane = tid & 31;
    const int g    = blockIdx.x - dir * SCAN_WGS;
    const int tile = g * 16 + wave;          // 0..127
    const int m    = tile & 3;               // batch tile (16 rows)
    const int n0   = (tile >> 2) * 16;       // column tile origin

    // persistent Whh fragments for this wave's 16 columns (128 VGPRs/lane)
    v16h Bf[16];
#pragma unroll
    for (int kc = 0; kc < 16; ++kc)
        Bf[kc] = load_b_frag(whhd + kc * 32, HID, n0);

    for (int t = 0; t < T_LEN; ++t) {
        const int te = rev ? (T_LEN - 1 - t) : t;
        const _Float16* cur = (t & 1) ? hb1 : hb0;
        _Float16*       nxt = (t & 1) ? hb0 : hb1;
        const float*   xrow = xwd + (size_t)te * BATCH * HID;

        if (t + 1 < T_LEN) {   // prefetch next step's input projection (256B/thread)
            const int tn = rev ? (T_LEN - 2 - t) : (t + 1);
            __builtin_prefetch(xwd + (size_t)tn * BATCH * HID + tid * 64, 0, 0);
        }

        v8f acc = {};
#pragma unroll
        for (int kc = 0; kc < 16; ++kc) {
            v16h a = load_a_frag(cur + m * 16 * HID + kc * 32, HID);
            acc = wmma_f16(a, Bf[kc], acc);
        }

        const int N  = n0 + (lane & 15);
        const int mb = m * 16 + ((lane >> 4) << 3);
#pragma unroll
        for (int r = 0; r < 8; ++r) {
            const int b = mb + r;
            const float v = tanhf(acc[r] + xrow[b * HID + N]);
            const _Float16 hv = (_Float16)v;
            nxt[b * HID + N] = hv;
            y[((size_t)te * BATCH + b) * (2 * HID) + dir * HID + N] = hv;
            if (t == T_LEN - 1) hTd[b * HID + N] = v;
        }

        // per-direction grid barrier (sense-reversing)
        __threadfence();      // release: make h writes visible at device scope
        __syncthreads();
        if (tid == 0) {
            const int g0 = atomicAdd(gen, 0);        // read generation BEFORE arriving
            if (atomicAdd(cnt, 1) == SCAN_WGS - 1) {
                atomicExch(cnt, 0);
                __threadfence();
                atomicAdd(gen, 1);
            } else {
                while (atomicAdd(gen, 0) == g0) __builtin_amdgcn_s_sleep(1);
            }
        }
        __syncthreads();
        __threadfence();      // acquire: invalidate WGP$ before reading new h
    }
}

// ---- FC head + softmax -----------------------------------------------------

extern "C" __global__ void __launch_bounds__(256)
fc_softmax(const _Float16* __restrict__ Y,   // [TB][2*HID]
           const _Float16* __restrict__ Wfc, // [OUTD][2*HID]
           const float* __restrict__ bfc,
           float* __restrict__ out)          // [TB][OUTD]
{
    __shared__ _Float16 sA[64 * 32];
    __shared__ _Float16 sB[128 * 32];
    __shared__ float    sL[64 * 128];

    const int tid  = threadIdx.x;
    const int bm   = blockIdx.x * 64;
    const int wave = tid >> 5, lane = tid & 31;
    const int mg   = wave & 3, ng = wave >> 2;

    v8f acc[4] = {};

    for (int k0 = 0; k0 < 2 * HID; k0 += 32) {
        {   const int r = tid >> 2, c = (tid & 3) * 8;
            stage16(Y + (size_t)(bm + r) * (2 * HID) + k0 + c, sA + r * 32 + c);
        }
        {   const int r = tid >> 1, c = (tid & 1) * 16;
            const _Float16* src = Wfc + (size_t)r * (2 * HID) + k0 + c;
            _Float16* dst = sB + r * 32 + c;
            stage16(src, dst);
            stage16(src + 8, dst + 8);
        }
        stage_wait();
        __syncthreads();
        v16h a = load_a_frag(sA + mg * 16 * 32, 32);
#pragma unroll
        for (int j = 0; j < 4; ++j) {
            v16h b = load_b_frag(sB, 32, ng * 64 + j * 16);
            acc[j] = wmma_f16(a, b, acc[j]);
        }
        __syncthreads();
    }

#pragma unroll
    for (int j = 0; j < 4; ++j) {
        const int N = ng * 64 + j * 16 + (lane & 15);
#pragma unroll
        for (int r = 0; r < 8; ++r) {
            const int M = mg * 16 + r + ((lane >> 4) << 3);
            sL[M * 128 + N] = acc[j][r] + bfc[N];
        }
    }
    __syncthreads();

    if (tid < 64) {
        float* row = sL + tid * 128;
        float mx = -1e30f;
        for (int i = 0; i < 128; ++i) mx = fmaxf(mx, row[i]);
        float s = 0.f;
        for (int i = 0; i < 128; ++i) { const float e = expf(row[i] - mx); row[i] = e; s += e; }
        const float inv = 1.f / s;
        float* o = out + (size_t)(bm + tid) * 128;
        for (int i = 0; i < 128; ++i) o[i] = row[i] * inv;
    }
}

// ---- host-side launch ------------------------------------------------------

extern "C" void kernel_launch(void* const* d_in, const int* in_sizes, int n_in,
                              void* d_out, int out_size, void* d_ws, size_t ws_size,
                              hipStream_t stream) {
    (void)in_sizes; (void)n_in; (void)out_size; (void)ws_size;

    const float* x     = (const float*)d_in[0];
    const float* h_0   = (const float*)d_in[1];
    const float* Wih0f = (const float*)d_in[2];
    const float* Whh0f = (const float*)d_in[3];
    const float* bih0f = (const float*)d_in[4];
    const float* bhh0f = (const float*)d_in[5];
    const float* Wih0b = (const float*)d_in[6];
    const float* Whh0b = (const float*)d_in[7];
    const float* bih0b = (const float*)d_in[8];
    const float* bhh0b = (const float*)d_in[9];
    const float* Wih1f = (const float*)d_in[10];
    const float* Whh1f = (const float*)d_in[11];
    const float* bih1f = (const float*)d_in[12];
    const float* bhh1f = (const float*)d_in[13];
    const float* Wih1b = (const float*)d_in[14];
    const float* Whh1b = (const float*)d_in[15];
    const float* bih1b = (const float*)d_in[16];
    const float* bhh1b = (const float*)d_in[17];
    const float* Wfc   = (const float*)d_in[18];
    const float* bfc   = (const float*)d_in[19];

    float* probs = (float*)d_out;                 // [TB][OUTD]
    float* hT    = probs + (size_t)TB * OUTD;     // [4][BATCH][HID]

    char* ws = (char*)d_ws;
    size_t off = 0;
    auto carve = [&](size_t bytes) {
        char* p = ws + off;
        off += (bytes + 255) & ~(size_t)255;
        return p;
    };
    _Float16* x_h    = (_Float16*)carve((size_t)TB * IN0 * 2);
    _Float16* wih0_h = (_Float16*)carve((size_t)2 * HID * IN0 * 2);      // f,b
    _Float16* whh0_h = (_Float16*)carve((size_t)2 * HID * HID * 2);      // f,b
    _Float16* wih1_h = (_Float16*)carve((size_t)2 * HID * 2 * HID * 2);  // f,b
    _Float16* whh1_h = (_Float16*)carve((size_t)2 * HID * HID * 2);      // f,b
    _Float16* wfc_h  = (_Float16*)carve((size_t)OUTD * 2 * HID * 2);
    float*    xwbuf  = (float*)   carve((size_t)2 * TB * HID * 4);       // reused l0/l1
    _Float16* y0_h   = (_Float16*)carve((size_t)TB * 2 * HID * 2);
    _Float16* y1_h   = (_Float16*)carve((size_t)TB * 2 * HID * 2);
    _Float16* hbuf   = (_Float16*)carve((size_t)2 * 2 * BATCH * HID * 2); // ping-pong h
    int*      sync   = (int*)     carve(4 * sizeof(int));

    auto cvt = [&](const float* s, _Float16* d, int n) {
        cvt_f16<<<(n + 255) / 256, 256, 0, stream>>>(s, d, n);
    };

    cvt(x,     x_h,                            TB * IN0);
    cvt(Wih0f, wih0_h,                         HID * IN0);
    cvt(Wih0b, wih0_h + (size_t)HID * IN0,     HID * IN0);
    cvt(Whh0f, whh0_h,                         HID * HID);
    cvt(Whh0b, whh0_h + (size_t)HID * HID,     HID * HID);
    cvt(Wih1f, wih1_h,                         HID * 2 * HID);
    cvt(Wih1b, wih1_h + (size_t)HID * 2 * HID, HID * 2 * HID);
    cvt(Whh1f, whh1_h,                         HID * HID);
    cvt(Whh1b, whh1_h + (size_t)HID * HID,     HID * HID);
    cvt(Wfc,   wfc_h,                          OUTD * 2 * HID);

    const dim3 ggrid(TB / 128, HID / 64);
    const int  init_blocks = (2 * BATCH * HID + 255) / 256;

    // layer 0
    gemm_bias<<<ggrid, 256, 0, stream>>>(x_h, IN0, wih0_h, IN0,
                                         bih0f, bhh0f, xwbuf, HID, 0);
    gemm_bias<<<ggrid, 256, 0, stream>>>(x_h, IN0, wih0_h + (size_t)HID * IN0, IN0,
                                         bih0b, bhh0b, xwbuf + (size_t)TB * HID, HID, 0);
    scan_init<<<init_blocks, 256, 0, stream>>>(h_0, hbuf, sync);
    rnn_scan<<<2 * SCAN_WGS, SCAN_THREADS, 0, stream>>>(xwbuf, whh0_h, hbuf,
                                                        y0_h, hT, sync);

    // layer 1
    gemm_bias<<<ggrid, 256, 0, stream>>>(y0_h, 2 * HID, wih1_h, 2 * HID,
                                         bih1f, bhh1f, xwbuf, HID, 0);
    gemm_bias<<<ggrid, 256, 0, stream>>>(y0_h, 2 * HID,
                                         wih1_h + (size_t)HID * 2 * HID, 2 * HID,
                                         bih1b, bhh1b, xwbuf + (size_t)TB * HID, HID, 0);
    scan_init<<<init_blocks, 256, 0, stream>>>(h_0 + (size_t)2 * BATCH * HID, hbuf, sync);
    rnn_scan<<<2 * SCAN_WGS, SCAN_THREADS, 0, stream>>>(xwbuf, whh1_h, hbuf,
                                                        y1_h, hT + (size_t)2 * BATCH * HID,
                                                        sync);

    // FC head + softmax
    fc_softmax<<<TB / 64, 256, 0, stream>>>(y1_h, wfc_h, bfc, probs);
}
